// SpatialSparseAttention_60206851555802
// MI455X (gfx1250) — matrix operand order repaired and checked by reading, not compiled, not run on hardware
//
#include <hip/hip_runtime.h>
#include <math.h>

#define HID   512
#define HQ    8
#define HKV   8
#define DD    64
#define CBS   4
#define SBS   8
#define TOPKK 8
#define WINSZ 8
#define SHIFTW 4
#define RESO  64
#define NB    16
#define MBLK  4096
#define NSS   8
#define SBLK  512
#define NWIN  9
#define NEGF  (-1e30f)

typedef __attribute__((ext_vector_type(16))) _Float16 v16h;
typedef __attribute__((ext_vector_type(8)))  _Float16 v8h;
typedef __attribute__((ext_vector_type(8)))  float    v8f;

static __device__ __forceinline__ int lane_id() { return threadIdx.x & 31; }

static __device__ __forceinline__ v8f wmma_f16(v16h a, v16h b, v8f c) {
  return __builtin_amdgcn_wmma_f32_16x16x32_f16(false, a, false, b, (short)0, c, false, false);
}

// ---- WMMA f16 fragment helpers (CDNA5 ISA 7.12.2 layouts, wave32) ----
// A fragment 16x32 from row-major f16 [*, lda]: per lane two contiguous 16B loads.
static __device__ __forceinline__ v16h frag_a_h(const _Float16* A, int lda, int row0, int k0) {
  int l = lane_id();
  const _Float16* rp = A + (size_t)(row0 + (l & 15)) * lda + k0 + ((l >> 4) << 3);
  v8h lo = *(const v8h*)rp;        // K = base+0..7
  v8h hi = *(const v8h*)(rp + 16); // K = base+16..23
  v16h a;
#pragma unroll
  for (int e = 0; e < 8; ++e) { a[e] = lo[e]; a[e + 8] = hi[e]; }
  return a;
}
// B fragment 32x16 where logical B[k][c] = X[c][k] (X row-major f16 [Nc,K]):
// per lane two 16B loads along K: base+0..7 and base+16..23 (base = 8*(lane/16)).
static __device__ __forceinline__ v16h frag_bt_h(const _Float16* X, int ldx, int k0, int col0) {
  int l = lane_id();
  const _Float16* cp = X + (size_t)(col0 + (l & 15)) * ldx + k0 + ((l >> 4) << 3);
  v8h lo = *(const v8h*)cp;
  v8h hi = *(const v8h*)(cp + 16);
  v16h b;
#pragma unroll
  for (int e = 0; e < 8; ++e) { b[e] = lo[e]; b[e + 8] = hi[e]; }
  return b;
}
// C/D layout: VGPR i -> row row0 + i + 8*(lane/16), col col0 + lane%16.
static __device__ __forceinline__ void frag_c_store(float* C, int ldc, int row0, int col0, v8f acc) {
  int l = lane_id();
  float* p = C + (size_t)(row0 + ((l >> 4) << 3)) * ldc + col0 + (l & 15);
#pragma unroll
  for (int i = 0; i < 8; ++i) p[(size_t)i * ldc] = acc[i];
}
static __device__ __forceinline__ void frag_c_store_h(_Float16* C, int ldc, int row0, int col0, v8f acc) {
  int l = lane_id();
  _Float16* p = C + (size_t)(row0 + ((l >> 4) << 3)) * ldc + col0 + (l & 15);
#pragma unroll
  for (int i = 0; i < 8; ++i) p[(size_t)i * ldc] = (_Float16)acc[i];
}
static __device__ __forceinline__ void frag_c_store_lds(float* S, int lds_, v8f acc, float scale) {
  int l = lane_id();
  float* p = S + (size_t)((l >> 4) << 3) * lds_ + (l & 15);
#pragma unroll
  for (int i = 0; i < 8; ++i) p[(size_t)i * lds_] = acc[i] * scale;
}

// ---- generic WMMA GEMM: C[Mr,Nc] = A[Mr,K] x BT[Nc,K]^T, one wave per 16x16 tile ----
__global__ void wmma_gemm_h(const _Float16* __restrict__ A, const _Float16* __restrict__ BT,
                            float* __restrict__ Cf, _Float16* __restrict__ Ch,
                            int Mr, int Nc, int K) {
  int tiles_n = Nc >> 4;
  int tile = blockIdx.x * (blockDim.x >> 5) + (threadIdx.x >> 5);
  int total = (Mr >> 4) * tiles_n;
  if (tile >= total) return;               // wave-uniform branch, EXEC all-1s for WMMA
  int tr = tile / tiles_n, tc = tile - tr * tiles_n;
  v8f acc = {};
  for (int k0 = 0; k0 < K; k0 += 32) {
    v16h a = frag_a_h(A, K, tr << 4, k0);
    v16h b = frag_bt_h(BT, K, k0, tc << 4);
    acc = wmma_f16(a, b, acc);
  }
  if (Cf) frag_c_store(Cf, Nc, tr << 4, tc << 4, acc);
  if (Ch) frag_c_store_h(Ch, Nc, tr << 4, tc << 4, acc);
}

// ---- conversions / transposes ----
__global__ void cvt_f16(const float* __restrict__ src, _Float16* __restrict__ dst, int n) {
  int i = blockIdx.x * blockDim.x + threadIdx.x;
  if (i < n) dst[i] = (_Float16)src[i];
}
// src f32 [R,C] row-major -> dst f16 [C,R] row-major
__global__ void transpose_cvt(const float* __restrict__ src, _Float16* __restrict__ dst,
                              int R, int C) {
  int i = blockIdx.x * blockDim.x + threadIdx.x;
  if (i >= R * C) return;
  int r = i / C, c = i - r * C;
  dst[(size_t)c * R + r] = (_Float16)src[i];
}

// ---- coordinate / id precompute ----
__global__ void prep_coords(const int* __restrict__ cf, int N, int* __restrict__ bid,
                            int* __restrict__ sbid, int* __restrict__ wid, int* __restrict__ peoff) {
  int n = blockIdx.x * blockDim.x + threadIdx.x;
  if (n >= N) return;
  int c = cf[n];
  int x = c / (RESO * RESO), y = (c / RESO) % RESO, z = c % RESO;
  bid[n]  = ((x / CBS) * NB + (y / CBS)) * NB + (z / CBS);
  sbid[n] = ((x / SBS) * NSS + (y / SBS)) * NSS + (z / SBS);
  int wx = (x + SHIFTW) / WINSZ, wy = (y + SHIFTW) / WINSZ, wz = (z + SHIFTW) / WINSZ;
  wid[n] = (wx * NWIN + wy) * NWIN + wz;
  peoff[n] = (((x % CBS) * CBS + (y % CBS)) * CBS + (z % CBS)) * HID;
}

__global__ void prep_sidm(int* __restrict__ sidm) {
  int m = blockIdx.x * blockDim.x + threadIdx.x;
  if (m >= MBLK) return;
  int mx = m / (NB * NB), my = (m / NB) % NB, mz = m % NB;
  sidm[m] = ((mx >> 1) * NSS + (my >> 1)) * NSS + (mz >> 1);
}

// ---- segment-sum of kf+pe and vf into compression blocks ----
__global__ void scatter_blocks(const float* __restrict__ kf, const float* __restrict__ vf,
                               const float* __restrict__ pe, const int* __restrict__ bid,
                               const int* __restrict__ peoff, float* __restrict__ sumK,
                               float* __restrict__ sumV, float* __restrict__ cnt) {
  int n = blockIdx.x;
  int b = bid[n];
  const float* pep = pe + peoff[n];
  for (int d = threadIdx.x; d < HID; d += blockDim.x) {
    atomicAdd(&sumK[(size_t)b * HID + d], kf[(size_t)n * HID + d] + pep[d]);
    atomicAdd(&sumV[(size_t)b * HID + d], vf[(size_t)n * HID + d]);
  }
  if (threadIdx.x == 0) atomicAdd(&cnt[b], 1.0f);
}

// mean-pool to f16 operands for the ck/cv projections
__global__ void normalize_blocks(const float* __restrict__ sumK, const float* __restrict__ sumV,
                                 const float* __restrict__ cnt, _Float16* __restrict__ tK,
                                 _Float16* __restrict__ tV) {
  int i = blockIdx.x * blockDim.x + threadIdx.x;
  if (i >= MBLK * HID) return;
  float d = fmaxf(cnt[i / HID], 1.0f);
  tK[i] = (_Float16)(sumK[i] / d);
  tV[i] = (_Float16)(sumV[i] / d);
}

// ---- mask evaluation (mode 0: compressed/valid, 1: top-k selection, 2: shifted window) ----
static __device__ __forceinline__ bool key_ok(int mode, int nrow, int key,
                                              const float* __restrict__ cnt,
                                              const int* __restrict__ topk,
                                              const int* __restrict__ sbid,
                                              const int* __restrict__ wid) {
  if (mode == 0) return cnt[key] > 0.0f;
  if (mode == 1) {
    int sb = sbid[key];
    bool ok = false;
#pragma unroll
    for (int t = 0; t < TOPKK; ++t) ok |= (topk[nrow * TOPKK + t] == sb);
    return ok;
  }
  return wid[nrow] == wid[key];
}

// ---- attention pass A: per (head, 16-query tile) streaming logsumexp ----
__global__ void attn_lse(const _Float16* __restrict__ Q, const _Float16* __restrict__ Kmat,
                         int Mkeys, int Nq, int mode,
                         const float* __restrict__ cnt, const int* __restrict__ topk,
                         const int* __restrict__ sbid, const int* __restrict__ wid,
                         float* __restrict__ lse) {
  __shared__ float sc[16 * 32];
  __shared__ float rm[16], rl[16];
  int h = blockIdx.x, n0 = blockIdx.y << 4;
  int l = lane_id();
  if (l < 16) { rm[l] = NEGF; rl[l] = 0.0f; }
  const _Float16* Qh = Q + h * DD;
  const _Float16* Kh = Kmat + h * DD;
  v16h a0 = frag_a_h(Qh, HID, n0, 0);    // hoisted: Q fragments reused for all chunks
  v16h a1 = frag_a_h(Qh, HID, n0, 32);
  for (int j0 = 0; j0 < Mkeys; j0 += 32) {
    v8f s0 = {}, s1 = {};
    s0 = wmma_f16(a0, frag_bt_h(Kh, HID, 0, j0), s0);
    s0 = wmma_f16(a1, frag_bt_h(Kh, HID, 32, j0), s0);
    s1 = wmma_f16(a0, frag_bt_h(Kh, HID, 0, j0 + 16), s1);
    s1 = wmma_f16(a1, frag_bt_h(Kh, HID, 32, j0 + 16), s1);
    frag_c_store_lds(sc, 32, s0, 0.125f);
    frag_c_store_lds(sc + 16, 32, s1, 0.125f);
    __syncthreads();
    if (l < 16) {
      int nrow = n0 + l;
      float mnew = rm[l];
      for (int j = 0; j < 32; ++j) {
        float v = key_ok(mode, nrow, j0 + j, cnt, topk, sbid, wid) ? sc[l * 32 + j] : NEGF;
        sc[l * 32 + j] = v;
        mnew = fmaxf(mnew, v);
      }
      float s = rl[l] * __expf(rm[l] - mnew);
      for (int j = 0; j < 32; ++j) s += __expf(sc[l * 32 + j] - mnew);
      rm[l] = mnew;
      rl[l] = s;
    }
    __syncthreads();
  }
  if (l < 16) lse[(size_t)h * Nq + n0 + l] = rm[l] + __logf(rl[l]);
}

// ---- attention pass B: exact softmax output (+ selection-block scores for mode 0) ----
// VT is the transposed value matrix, f16 [HID, ldvt].
__global__ void attn_out(const _Float16* __restrict__ Q, const _Float16* __restrict__ Kmat,
                         const _Float16* __restrict__ VT, int ldvt, int Mkeys, int Nq, int mode,
                         const float* __restrict__ cnt, const int* __restrict__ topk,
                         const int* __restrict__ sbid, const int* __restrict__ wid,
                         const int* __restrict__ sidm, const float* __restrict__ lse,
                         float* __restrict__ out, float* __restrict__ sel_score) {
  __shared__ float sc[16 * 32];
  __shared__ __align__(16) _Float16 ph[16 * 32];
  __shared__ float rlse[16];
  __shared__ float selbuf[16 * SBLK];
  int h = blockIdx.x, n0 = blockIdx.y << 4;
  int l = lane_id();
  if (mode == 0)
    for (int i = l; i < 16 * SBLK; i += 32) selbuf[i] = 0.0f;
  if (l < 16) rlse[l] = lse[(size_t)h * Nq + n0 + l];
  __syncthreads();
  const _Float16* Qh = Q + h * DD;
  const _Float16* Kh = Kmat + h * DD;
  const _Float16* V0 = VT + (size_t)(h * DD + 0)  * ldvt;
  const _Float16* V1 = VT + (size_t)(h * DD + 16) * ldvt;
  const _Float16* V2 = VT + (size_t)(h * DD + 32) * ldvt;
  const _Float16* V3 = VT + (size_t)(h * DD + 48) * ldvt;
  v16h a0 = frag_a_h(Qh, HID, n0, 0);
  v16h a1 = frag_a_h(Qh, HID, n0, 32);
  v8f o0 = {}, o1 = {}, o2 = {}, o3 = {};
  for (int j0 = 0; j0 < Mkeys; j0 += 32) {
    v8f s0 = {}, s1 = {};
    s0 = wmma_f16(a0, frag_bt_h(Kh, HID, 0, j0), s0);
    s0 = wmma_f16(a1, frag_bt_h(Kh, HID, 32, j0), s0);
    s1 = wmma_f16(a0, frag_bt_h(Kh, HID, 0, j0 + 16), s1);
    s1 = wmma_f16(a1, frag_bt_h(Kh, HID, 32, j0 + 16), s1);
    frag_c_store_lds(sc, 32, s0, 0.125f);
    frag_c_store_lds(sc + 16, 32, s1, 0.125f);
    __syncthreads();
    // lane l owns column l; p = exp(s - lse) stored as f16 for the PV A-fragment
    {
      int key = j0 + l;
      int sid = (mode == 0) ? sidm[key] : 0;
      for (int r = 0; r < 16; ++r) {
        float v = key_ok(mode, n0 + r, key, cnt, topk, sbid, wid) ? sc[r * 32 + l] : NEGF;
        float p = __expf(v - rlse[r]);
        ph[r * 32 + l] = (_Float16)p;
        if (mode == 0) atomicAdd(&selbuf[r * SBLK + sid], p);
      }
    }
    __syncthreads();
    v16h pa = frag_a_h(ph, 32, 0, 0);
    o0 = wmma_f16(pa, frag_bt_h(V0, ldvt, j0, 0), o0);
    o1 = wmma_f16(pa, frag_bt_h(V1, ldvt, j0, 0), o1);
    o2 = wmma_f16(pa, frag_bt_h(V2, ldvt, j0, 0), o2);
    o3 = wmma_f16(pa, frag_bt_h(V3, ldvt, j0, 0), o3);
    __syncthreads();
  }
  frag_c_store(out + h * DD, HID, n0, 0, o0);
  frag_c_store(out + h * DD, HID, n0, 16, o1);
  frag_c_store(out + h * DD, HID, n0, 32, o2);
  frag_c_store(out + h * DD, HID, n0, 48, o3);
  if (mode == 0) {
    __syncthreads();
    for (int i = l; i < 16 * SBLK; i += 32) {
      float v = selbuf[i];
      if (v != 0.0f)
        atomicAdd(&sel_score[(size_t)(n0 + i / SBLK) * SBLK + (i % SBLK)], v);
    }
  }
}

// ---- top-k selection blocks per token (stable, ties -> lowest index first) ----
__global__ void topk_kernel(const float* __restrict__ sel, int* __restrict__ topk) {
  __shared__ float s[SBLK];
  int n = blockIdx.x;
  for (int i = threadIdx.x; i < SBLK; i += blockDim.x) s[i] = sel[(size_t)n * SBLK + i];
  __syncthreads();
  if (threadIdx.x == 0) {
    for (int t = 0; t < TOPKK; ++t) {
      float best = -INFINITY;
      int bi = 0;
      for (int i = 0; i < SBLK; ++i) {
        float v = s[i];
        if (v > best) { best = v; bi = i; }
      }
      topk[n * TOPKK + t] = bi;
      s[bi] = -INFINITY;
    }
  }
}

// ---- gating ----
__global__ void gate_kernel(const float* __restrict__ feats, const float* __restrict__ Wg,
                            float* __restrict__ gate, int N) {
  int n = blockIdx.x * blockDim.x + threadIdx.x;
  if (n >= N) return;
  float a0 = 0.f, a1 = 0.f, a2 = 0.f;
  const float* f = feats + (size_t)n * HID;
  for (int k = 0; k < HID; ++k) {
    float x = f[k];
    a0 += x * Wg[k * 3 + 0];
    a1 += x * Wg[k * 3 + 1];
    a2 += x * Wg[k * 3 + 2];
  }
  gate[n * 3 + 0] = 1.f / (1.f + __expf(-a0));
  gate[n * 3 + 1] = 1.f / (1.f + __expf(-a1));
  gate[n * 3 + 2] = 1.f / (1.f + __expf(-a2));
}

__global__ void fuse_kernel(const float* __restrict__ oc, const float* __restrict__ os,
                            const float* __restrict__ ow, const float* __restrict__ gate,
                            _Float16* __restrict__ fused, int N) {
  int i = blockIdx.x * blockDim.x + threadIdx.x;
  if (i >= N * HID) return;
  int n = i / HID;
  fused[i] = (_Float16)(gate[n * 3 + 0] * oc[i] + gate[n * 3 + 1] * os[i] +
                        gate[n * 3 + 2] * ow[i]);
}

extern "C" void kernel_launch(void* const* d_in, const int* in_sizes, int n_in,
                              void* d_out, int out_size, void* d_ws, size_t ws_size,
                              hipStream_t stream) {
  (void)n_in; (void)out_size; (void)ws_size;
  const float* feats = (const float*)d_in[0];
  const int*   cf    = (const int*)d_in[1];
  const float* Wq    = (const float*)d_in[2];
  const float* Wk    = (const float*)d_in[3];
  const float* Wv    = (const float*)d_in[4];
  const float* Wo    = (const float*)d_in[5];
  const float* Wck   = (const float*)d_in[6];
  const float* Wcv   = (const float*)d_in[7];
  const float* pe    = (const float*)d_in[8];
  const float* Wg    = (const float*)d_in[9];
  float* outp = (float*)d_out;

  const int N = in_sizes[0] / HID;  // 2048
  const size_t NH = (size_t)N * HID, MH = (size_t)MBLK * HID, WW = (size_t)HID * HID;

  // ---- workspace layout: f32 region, then f16 region, then ints ----
  float* ws    = (float*)d_ws;
  float* kf    = ws;                    // [N,HID]  f32 (scatter source)
  float* vf    = kf + NH;               // [N,HID]  f32
  float* sumK  = vf + NH;               // [M,HID]
  float* sumV  = sumK + MH;             // [M,HID]
  float* cnt   = sumV + MH;             // [M]
  float* cvf   = cnt + MBLK;            // [M,HID]  f32 (pre-transpose cv)
  float* lse   = cvf + MH;              // [HQ,N]
  float* out_c = lse + (size_t)HQ * N;  // [N,HID]
  float* out_s = out_c + NH;            // [N,HID]
  float* out_w = out_s + NH;            // [N,HID]
  float* gate  = out_w + NH;            // [N,3]
  float* sel   = sumK;                  // alias: sumK dead after normalize_blocks

  _Float16* fp16  = (_Float16*)(gate + (size_t)N * 3 + 512);  // pad to keep 1KB alignment
  _Float16* feats_h = fp16;             // [N,HID]
  _Float16* WqT   = feats_h + NH;       // [HID,HID] transposed weights
  _Float16* WkT   = WqT + WW;
  _Float16* WvT   = WkT + WW;
  _Float16* WckT  = WvT + WW;
  _Float16* WcvT  = WckT + WW;
  _Float16* WoT   = WcvT + WW;
  _Float16* q_h   = WoT + WW;           // [N,HID]
  _Float16* kf_h  = q_h + NH;           // [N,HID]
  _Float16* vfT   = kf_h + NH;          // [HID,N]
  _Float16* tK_h  = vfT + NH;           // [M,HID]
  _Float16* tV_h  = tK_h + MH;          // [M,HID]
  _Float16* ck_h  = tV_h + MH;          // [M,HID]
  _Float16* cvT   = ck_h + MH;          // [HID,M]
  _Float16* fus_h = cvT + MH;           // [N,HID]

  int* bid   = (int*)(fus_h + NH);
  int* sbid  = bid + N;
  int* wid   = sbid + N;
  int* peoff = wid + N;
  int* topk  = peoff + N;               // [N,8]
  int* sidm  = topk + (size_t)N * TOPKK;// [M]

  hipMemsetAsync(sumK, 0, (2 * MH + MBLK) * sizeof(float), stream);

  prep_coords<<<(N + 255) / 256, 256, 0, stream>>>(cf, N, bid, sbid, wid, peoff);
  prep_sidm<<<(MBLK + 255) / 256, 256, 0, stream>>>(sidm);

  // one-time f16 conversions (weights transposed so B fragments are contiguous along K)
  cvt_f16<<<((int)NH + 255) / 256, 256, 0, stream>>>(feats, feats_h, (int)NH);
  transpose_cvt<<<((int)WW + 255) / 256, 256, 0, stream>>>(Wq, WqT, HID, HID);
  transpose_cvt<<<((int)WW + 255) / 256, 256, 0, stream>>>(Wk, WkT, HID, HID);
  transpose_cvt<<<((int)WW + 255) / 256, 256, 0, stream>>>(Wv, WvT, HID, HID);
  transpose_cvt<<<((int)WW + 255) / 256, 256, 0, stream>>>(Wck, WckT, HID, HID);
  transpose_cvt<<<((int)WW + 255) / 256, 256, 0, stream>>>(Wcv, WcvT, HID, HID);
  transpose_cvt<<<((int)WW + 255) / 256, 256, 0, stream>>>(Wo, WoT, HID, HID);

  // q/k/v projections (f16 in, f32+f16 out as needed)
  int tilesN = (N / 16) * (HID / 16);
  wmma_gemm_h<<<(tilesN + 7) / 8, 256, 0, stream>>>(feats_h, WqT, nullptr, q_h, N, HID, HID);
  wmma_gemm_h<<<(tilesN + 7) / 8, 256, 0, stream>>>(feats_h, WkT, kf, kf_h, N, HID, HID);
  wmma_gemm_h<<<(tilesN + 7) / 8, 256, 0, stream>>>(feats_h, WvT, vf, nullptr, N, HID, HID);
  transpose_cvt<<<((int)NH + 255) / 256, 256, 0, stream>>>(vf, vfT, N, HID);

  // block compression: mean-pool (with key PE) then project
  scatter_blocks<<<N, 128, 0, stream>>>(kf, vf, pe, bid, peoff, sumK, sumV, cnt);
  normalize_blocks<<<((int)MH + 255) / 256, 256, 0, stream>>>(sumK, sumV, cnt, tK_h, tV_h);
  int tilesM = (MBLK / 16) * (HID / 16);
  wmma_gemm_h<<<(tilesM + 7) / 8, 256, 0, stream>>>(tK_h, WckT, nullptr, ck_h, MBLK, HID, HID);
  wmma_gemm_h<<<(tilesM + 7) / 8, 256, 0, stream>>>(tV_h, WcvT, cvf, nullptr, MBLK, HID, HID);
  transpose_cvt<<<((int)MH + 255) / 256, 256, 0, stream>>>(cvf, cvT, MBLK, HID);

  // sel_score buffer (aliases sumK) becomes live only now
  hipMemsetAsync(sel, 0, (size_t)N * SBLK * sizeof(float), stream);

  dim3 agrid(HQ, N / 16);
  // compressed attention (mode 0) + selection-block scores
  attn_lse<<<agrid, 32, 0, stream>>>(q_h, ck_h, MBLK, N, 0, cnt, topk, sbid, wid, lse);
  attn_out<<<agrid, 32, 0, stream>>>(q_h, ck_h, cvT, MBLK, MBLK, N, 0, cnt, topk, sbid, wid,
                                     sidm, lse, out_c, sel);
  // top-k selection blocks
  topk_kernel<<<N, 32, 0, stream>>>(sel, topk);
  // selection attention (mode 1)
  attn_lse<<<agrid, 32, 0, stream>>>(q_h, kf_h, N, N, 1, cnt, topk, sbid, wid, lse);
  attn_out<<<agrid, 32, 0, stream>>>(q_h, kf_h, vfT, N, N, N, 1, cnt, topk, sbid, wid,
                                     sidm, lse, out_s, nullptr);
  // shifted-window attention (mode 2)
  attn_lse<<<agrid, 32, 0, stream>>>(q_h, kf_h, N, N, 2, cnt, topk, sbid, wid, lse);
  attn_out<<<agrid, 32, 0, stream>>>(q_h, kf_h, vfT, N, N, N, 2, cnt, topk, sbid, wid,
                                     sidm, lse, out_w, nullptr);

  // gated fusion + output projection
  gate_kernel<<<(N + 255) / 256, 256, 0, stream>>>(feats, Wg, gate, N);
  fuse_kernel<<<((int)NH + 255) / 256, 256, 0, stream>>>(out_c, out_s, out_w, gate, fus_h, N);
  wmma_gemm_h<<<(tilesN + 7) / 8, 256, 0, stream>>>(fus_h, WoT, outp, nullptr, N, HID, HID);
}
